// HSumGraph_4252017623524
// MI455X (gfx1250) — compile-verified, matrix operand order checked by reading, not run on hardware
//
#include <hip/hip_runtime.h>
#include <hip/hip_bf16.h>
#include <math.h>

typedef _Float16 h16;
typedef __attribute__((ext_vector_type(16))) _Float16 v16h;
typedef __attribute__((ext_vector_type(8)))  float    v8f;

#define WMMA32(a,b,c) __builtin_amdgcn_wmma_f32_16x16x32_f16(false,(a),false,(b),(short)0,(c),false,false)

#define NWORDS 80000
#define NSENTS 3200
#define NEDGES 320000
#define EDIM   300

__device__ __forceinline__ v8f vzero(){ v8f z; 
#pragma unroll
  for(int i=0;i<8;++i) z[i]=0.f; return z; }

// Fragment loader for A (row-major [M][lda]) and Bt ([N][ldb]) tiles.
// CDNA5 16x32 f16 layout: lane<16 -> row lane, K {0..7,16..23}; lane>=16 -> row lane-16, K {8..15,24..31}
__device__ __forceinline__ v16h ldfrag(const h16* p0, int stride){
  int lane = threadIdx.x & 31;
  const h16* p = p0 + (size_t)(lane & 15) * stride + ((lane >> 4) << 3);
  v16h r;
#pragma unroll
  for (int i = 0; i < 8; ++i){ r[i] = p[i]; r[i+8] = p[i+16]; }
  return r;
}

__device__ __forceinline__ float pos_tab(int pos, int i){
  float ang = (float)pos / powf(10000.f, (2.f*(float)(i>>1))/300.f);
  return (i & 1) ? cosf(ang) : sinf(ang);
}

__device__ __forceinline__ unsigned fkey(float f){
  unsigned b=__float_as_uint(f); return (b&0x80000000u)?~b:(b|0x80000000u);
}
__device__ __forceinline__ float funkey(unsigned k){
  unsigned b=(k&0x80000000u)?(k^0x80000000u):~k; return __uint_as_float(b);
}

// ---------------- generic WMMA GEMM: C[M][ldc] (+f16 Ch) = A[M][lda] * Bt[N][ldb]^T + bias (+relu) (+res)
__global__ void __launch_bounds__(256) gemm_h(
  const h16* __restrict__ A, int lda, const h16* __restrict__ Bt, int ldb,
  float* __restrict__ C, int ldc, h16* __restrict__ Ch, int ldh,
  const float* __restrict__ bias, const float* __restrict__ res,
  int M, int N, int Ksteps, int relu)
{
  int wave = threadIdx.x >> 5, lane = threadIdx.x & 31;
  int m0 = blockIdx.y*128 + wave*16;
  int n0 = blockIdx.x*64;
  if (m0 >= M) return;
  v8f acc[4]; 
#pragma unroll
  for (int t=0;t<4;++t) acc[t]=vzero();
  for (int ks=0; ks<Ksteps; ++ks){
    int k0 = ks*32;
    int m = m0 + (lane & 15); if (m > M-1) m = M-1;
    const h16* pa = A + (size_t)m*lda + k0 + ((lane>>4)<<3);
    v16h a;
#pragma unroll
    for (int i=0;i<8;++i){ a[i]=pa[i]; a[i+8]=pa[i+16]; }
#pragma unroll
    for (int t=0;t<4;++t){
      v16h b = ldfrag(Bt + (size_t)(n0 + t*16)*ldb + k0, ldb);
      acc[t] = WMMA32(a,b,acc[t]);
    }
  }
  int col = lane & 15, rb = (lane>>4)<<3;
#pragma unroll
  for (int j=0;j<8;++j){
    int m = m0 + rb + j;
    if (m >= M) continue;
#pragma unroll
    for (int t=0;t<4;++t){
      int n = n0 + t*16 + col;
      if (n >= N) continue;
      float v = acc[t][j];
      if (bias) v += bias[n];
      if (relu) v = v>0.f ? v : 0.f;
      if (res)  v += res[(size_t)m*ldc + n];
      if (C)  C[(size_t)m*ldc + n] = v;
      if (Ch) Ch[(size_t)m*ldh + n] = (h16)v;
    }
  }
}

// ---------------- tok build: embed gather + sinusoid pos, f16, zero tail pad
__global__ void build_tok(const int* __restrict__ sw, const float* __restrict__ embed,
                          h16* __restrict__ tok, long total){
  long idx = (long)blockIdx.x*blockDim.x + threadIdx.x;
  if (idx >= total) return;
  int row = (int)(idx/EDIM), e = (int)(idx%EDIM);
  float v = 0.f;
  if (row < NSENTS*100){
    int s = row/100, t = row%100;
    v = embed[(size_t)sw[s*100+t]*EDIM + e] + pos_tab(t+1, e);
  }
  tok[idx] = (h16)v;
}

__global__ void build_wordf(const int* __restrict__ wid, const float* __restrict__ embed,
                            h16* __restrict__ wfh, float* __restrict__ wstate){
  long idx = (long)blockIdx.x*blockDim.x + threadIdx.x;
  if (idx >= (long)NWORDS*320) return;
  int n = (int)(idx/320), d = (int)(idx%320);
  float v = (d<EDIM) ? embed[(size_t)wid[n]*EDIM + d] : 0.f;
  wfh[idx] = (h16)v;
  if (d<EDIM) wstate[(size_t)n*EDIM + d] = v;
}

// ---------------- fused im2col-free CNN + relu + max-over-t + pos-doc add
__global__ void __launch_bounds__(256) conv_ngram(
  const h16* __restrict__ tok, const h16* __restrict__ cw, const float* __restrict__ cb,
  const int* __restrict__ spos,
  h16* __restrict__ ngram_h, h16* __restrict__ ngramp_h)
{
  __shared__ unsigned smax[6*64];
  const int kst[6]  = {19,29,38,47,57,66};          // ceil(k*300/32)
  const int koff[6] = {0,38912,98304,176128,272384,389120};
  const int jcum[7] = {0,7,14,21,27,33,39};
  int s = blockIdx.x;
  for (int i=threadIdx.x;i<384;i+=256) smax[i]=0u;
  __syncthreads();
  int wave=threadIdx.x>>5, lane=threadIdx.x&31;
  int col=lane&15, rb=(lane>>4)<<3;
  const h16* ts = tok + (size_t)s*30000;
  for (int job=wave; job<39; job+=8){
    int ki=0; while (job >= jcum[ki+1]) ++ki;
    int tile = job - jcum[ki];
    int kwid = ki+2;
    int t0 = tile*16;
    int nt = 101 - kwid;
    const h16* Bt = cw + koff[ki];
    int ldb = kst[ki]*32;
    v8f acc[4];
#pragma unroll
    for (int t=0;t<4;++t) acc[t]=vzero();
    for (int ks2=0; ks2<kst[ki]; ++ks2){
      int k0 = ks2*32;
      const h16* pa = ts + (size_t)(t0 + (lane&15))*EDIM + k0 + ((lane>>4)<<3);
      v16h a;
#pragma unroll
      for (int i=0;i<8;++i){ a[i]=pa[i]; a[i+8]=pa[i+16]; }
#pragma unroll
      for (int t=0;t<4;++t){
        v16h b = ldfrag(Bt + (size_t)(t*16)*ldb + k0, ldb);
        acc[t] = WMMA32(a,b,acc[t]);
      }
    }
#pragma unroll
    for (int j=0;j<8;++j){
      int tpos = t0 + rb + j;
      if (tpos >= nt) continue;
#pragma unroll
      for (int t=0;t<4;++t){
        int c = t*16 + col;
        float v = acc[t][j] + cb[ki*64+c];
        v = v>0.f ? v : 0.f;
        atomicMax(&smax[ki*64+c], __float_as_uint(v));
      }
    }
  }
  __syncthreads();
  int pos = spos[s];
  for (int c=threadIdx.x; c<320; c+=256){
    float v = 0.f;
    if (c < EDIM) v = __uint_as_float(smax[(c/50)*64 + (c%50)]);
    ngram_h[(size_t)s*320 + c] = (h16)v;
    float p = (c<EDIM) ? pos_tab(pos, c) : 0.f;
    ngramp_h[(size_t)s*320 + c] = (h16)(v + p);
  }
}

// ---------------- LSTM scan: single workgroup, h kept f16 in LDS, h@Whh^T via WMMA
__global__ void __launch_bounds__(512) lstm_scan(
  const float* __restrict__ pre, const h16* __restrict__ WhhT,
  h16* __restrict__ outH, int ostride, int ooff, int reverse)
{
  __shared__ h16 hbuf[64*128];
  int wave=threadIdx.x>>5, lane=threadIdx.x&31;
  int col=lane&15, rb=(lane>>4)<<3;
  for (int i=threadIdx.x;i<64*128;i+=512) hbuf[i]=(h16)0.f;
  float cst[2][8];
#pragma unroll
  for (int jj=0;jj<2;++jj)
#pragma unroll
    for (int j=0;j<8;++j) cst[jj][j]=0.f;
  __syncthreads();
  for (int t=0;t<50;++t){
    int tt = reverse ? (49-t) : t;
    float hnew[2][8];
#pragma unroll
    for (int jj=0;jj<2;++jj){
      int job = wave*2+jj;
      int mt = job>>3, nb = job&7;
      v8f ai=vzero(), af=vzero(), ag=vzero(), ao=vzero();
#pragma unroll
      for (int j=0;j<8;++j){
        int b = mt*16 + rb + j;
        int n = nb*16 + col;
        size_t base = ((size_t)(b*50+tt))*512;
        ai[j]=pre[base+n]; af[j]=pre[base+128+n]; ag[j]=pre[base+256+n]; ao[j]=pre[base+384+n];
      }
#pragma unroll
      for (int ks=0;ks<4;++ks){
        int k0=ks*32;
        v16h a = ldfrag(hbuf + mt*16*128 + k0, 128);
        ai = WMMA32(a, ldfrag(WhhT + (size_t)(  0 + nb*16)*128 + k0, 128), ai);
        af = WMMA32(a, ldfrag(WhhT + (size_t)(128 + nb*16)*128 + k0, 128), af);
        ag = WMMA32(a, ldfrag(WhhT + (size_t)(256 + nb*16)*128 + k0, 128), ag);
        ao = WMMA32(a, ldfrag(WhhT + (size_t)(384 + nb*16)*128 + k0, 128), ao);
      }
#pragma unroll
      for (int j=0;j<8;++j){
        float iv = 1.f/(1.f+expf(-ai[j]));
        float fv = 1.f/(1.f+expf(-af[j]));
        float gv = tanhf(ag[j]);
        float ov = 1.f/(1.f+expf(-ao[j]));
        float c = fv*cst[jj][j] + iv*gv;
        cst[jj][j] = c;
        hnew[jj][j] = ov*tanhf(c);
      }
    }
    __syncthreads();
#pragma unroll
    for (int jj=0;jj<2;++jj){
      int job=wave*2+jj; int mt=job>>3, nb=job&7;
#pragma unroll
      for (int j=0;j<8;++j){
        int b = mt*16+rb+j, n = nb*16+col;
        h16 hh = (h16)hnew[jj][j];
        hbuf[b*128+n]=hh;
        outH[(size_t)(b*50+tt)*ostride + ooff + n] = hh;
      }
    }
    __syncthreads();
  }
}

// ---------------- param conversion kernels
__global__ void conv_trans(const float* __restrict__ W, h16* __restrict__ Bt,
                           int K, int N, int Kp, int Np){      // Bt[n][k] = W[k][n]
  long idx=(long)blockIdx.x*blockDim.x+threadIdx.x;
  if (idx>=(long)Np*Kp) return;
  int n=(int)(idx/Kp), k=(int)(idx%Kp);
  Bt[idx] = (n<N && k<K) ? (h16)W[(size_t)k*N+n] : (h16)0.f;
}
__global__ void copy_rowmajor(const float* __restrict__ W, h16* __restrict__ Bt,
                              int N, int K, int Kp, int Np){   // Bt[n][k] = W[n][k]
  long idx=(long)blockIdx.x*blockDim.x+threadIdx.x;
  if (idx>=(long)Np*Kp) return;
  int n=(int)(idx/Kp), k=(int)(idx%Kp);
  Bt[idx] = (n<N && k<K) ? (h16)W[(size_t)n*K+k] : (h16)0.f;
}
__global__ void fc_trans(const float* __restrict__ fc, h16* __restrict__ Bt,
                         int nh, int D, int dh, int Kp, int Np){ // Bt[h*dh+e][d]=fc[h][d][e]
  long idx=(long)blockIdx.x*blockDim.x+threadIdx.x;
  if (idx>=(long)Np*Kp) return;
  int n=(int)(idx/Kp), k=(int)(idx%Kp);
  float v=0.f;
  if (n<nh*dh && k<D){ int h=n/dh, e=n%dh; v = fc[((size_t)h*D+k)*dh+e]; }
  Bt[idx]=(h16)v;
}
__global__ void addb(const float* a, const float* b, float* c, int n){
  int i=blockIdx.x*blockDim.x+threadIdx.x; if(i<n) c[i]=a[i]+b[i];
}
__global__ void copy_pad_f(const float* s, float* d, int n, int np){
  int i=blockIdx.x*blockDim.x+threadIdx.x; if(i<np) d[i]= (i<n)? s[i] : 0.f;
}
__global__ void zero_u32(unsigned* __restrict__ p, long n){
  long i=(long)blockIdx.x*blockDim.x+threadIdx.x; if(i<n) p[i]=0u;
}
__global__ void gat_dfs(const float* __restrict__ tfe, const float* __restrict__ ffw,
                        const float* __restrict__ ffb, const float* __restrict__ afeat,
                        float* __restrict__ dfs, int nh, int dh){
  int i=threadIdx.x; if(i>=10*nh) return;
  int k=i/nh, h=i%nh;
  float s=0.f;
  for (int e=0;e<dh;++e){
    float de=ffb[h*dh+e];
    for (int f=0;f<10;++f) de += tfe[k*10+f]*ffw[((size_t)h*10+f)*dh+e];
    s += de*afeat[h*dh+e];
  }
  dfs[i]=s;
}

// ---------------- GAT edge / node kernels
__global__ void gat_salpha(const float* __restrict__ z, const float* __restrict__ asrc,
                           float* __restrict__ sal, int Nn, int nh, int dh){
  long i=(long)blockIdx.x*blockDim.x+threadIdx.x;
  if (i>=(long)Nn*nh) return;
  int n=(int)(i/nh), h=(int)(i%nh);
  int dim=nh*dh;
  float s=0.f;
  for (int d=0;d<dh;++d) s += z[(size_t)n*dim + h*dh + d]*asrc[h*dh+d];
  sal[i]=s;
}
__global__ void gat_e1(const int* __restrict__ esrc, const int* __restrict__ edst,
                       const int* __restrict__ tff, const float* __restrict__ sal,
                       const float* __restrict__ dfs, float* __restrict__ att,
                       unsigned* __restrict__ amax, int E, int nh){
  int e=blockIdx.x*blockDim.x+threadIdx.x; if(e>=E) return;
  int s=esrc[e], d=edst[e], k=tff[e];
  for (int h=0;h<nh;++h){
    float a = sal[(size_t)s*nh+h] + dfs[k*nh+h];
    a = a>0.f ? a : 0.01f*a;
    att[(size_t)e*nh+h]=a;
    atomicMax(&amax[(size_t)d*nh+h], fkey(a));
  }
}
__global__ void gat_e2(const int* __restrict__ edst, const float* __restrict__ att,
                       const unsigned* __restrict__ amax, float* __restrict__ ex,
                       float* __restrict__ den, int E, int nh){
  int e=blockIdx.x*blockDim.x+threadIdx.x; if(e>=E) return;
  int d=edst[e];
  for (int h=0;h<nh;++h){
    float m = funkey(amax[(size_t)d*nh+h]);
    float x = expf(att[(size_t)e*nh+h]-m);
    ex[(size_t)e*nh+h]=x;
    atomicAdd(&den[(size_t)d*nh+h], x);
  }
}
__global__ void gat_e3(const int* __restrict__ esrc, const int* __restrict__ edst,
                       const float* __restrict__ ex, const float* __restrict__ den,
                       const float* __restrict__ z, float* __restrict__ hsum,
                       int E, int nh, int dh, int dim){
  long idx=(long)blockIdx.x*blockDim.x+threadIdx.x;
  if (idx>=(long)E*dim) return;
  int e=(int)(idx/dim), dc=(int)(idx%dim);
  int h=dc/dh;
  int s=esrc[e], d=edst[e];
  float al = ex[(size_t)e*nh+h] / fmaxf(den[(size_t)d*nh+h], 1e-12f);
  atomicAdd(&hsum[(size_t)d*dim+dc], al*z[(size_t)s*dim+dc]);
}
// elu + residual (in place) + layernorm -> f16 (one wave per node)
__global__ void __launch_bounds__(256) gat_node(
  float* __restrict__ hsum, const float* __restrict__ origin,
  const float* __restrict__ lng, const float* __restrict__ lnb,
  h16* __restrict__ yh, int dim, int kpad, int nodes)
{
  int node = blockIdx.x*8 + (threadIdx.x>>5);
  int lane = threadIdx.x & 31;
  if (node >= nodes) return;
  float* row = hsum + (size_t)node*dim;
  const float* orow = origin + (size_t)node*dim;
  int cnt = (dim+31)/32;
  float vals[10];
  float lsum=0.f;
  for (int i=0;i<cnt;++i){
    int d=lane+32*i; float v=0.f;
    if (d<dim){ float x=row[d]; x = x>0.f ? x : (expf(x)-1.f); v = x + orow[d]; vals[i]=v; row[d]=v; }
    else vals[i]=0.f;
    lsum += v;
  }
  for (int m=16;m>=1;m>>=1) lsum += __shfl_xor(lsum, m, 32);
  float mean = lsum/(float)dim;
  float vs=0.f;
  for (int i=0;i<cnt;++i){ int d=lane+32*i; if(d<dim){ float dd=vals[i]-mean; vs+=dd*dd; } }
  for (int m=16;m>=1;m>>=1) vs += __shfl_xor(vs, m, 32);
  float rstd = rsqrtf(vs/(float)dim + 1e-6f);
  for (int i=0;i<cnt;++i){ int d=lane+32*i; if(d<dim)
      yh[(size_t)node*kpad+d] = (h16)((vals[i]-mean)*rstd*lng[d]+lnb[d]); }
  for (int d=dim+lane; d<kpad; d+=32) yh[(size_t)node*kpad+d]=(h16)0.f;
}

__global__ void final_out(const float* __restrict__ ss, const float* __restrict__ w,
                          const float* __restrict__ b, float* __restrict__ out){
  int i=blockIdx.x*blockDim.x+threadIdx.x;
  if (i>=NSENTS*2) return;
  int s=i>>1, j=i&1;
  float v=b[j];
  for (int d=0;d<64;++d) v += ss[(size_t)s*64+d]*w[d*2+j];
  out[i]=v;
}

extern "C" void kernel_launch(void* const* d_in, const int* in_sizes, int n_in,
                              void* d_out, int out_size, void* d_ws, size_t ws_size,
                              hipStream_t stream)
{
  if (n_in < 65) return;
  const int*   wid  = (const int*)d_in[0];
  const int*   sw   = (const int*)d_in[1];
  const int*   spos = (const int*)d_in[2];
  const int*   esrc = (const int*)d_in[3];
  const int*   edst = (const int*)d_in[4];
  const int*   tff  = (const int*)d_in[5];
  const float* embed= (const float*)d_in[6];
  const float* tfe  = (const float*)d_in[7];
  const float* cvw[6], *cvb[6];
  for (int i=0;i<6;++i){ cvw[i]=(const float*)d_in[8+2*i]; cvb[i]=(const float*)d_in[9+2*i]; }
  const float* cnnw=(const float*)d_in[20]; const float* cnnb=(const float*)d_in[21];
  const float* Wih[4]={(const float*)d_in[22],(const float*)d_in[26],(const float*)d_in[30],(const float*)d_in[34]};
  const float* Whh[4]={(const float*)d_in[23],(const float*)d_in[27],(const float*)d_in[31],(const float*)d_in[35]};
  const float* bih[4]={(const float*)d_in[24],(const float*)d_in[28],(const float*)d_in[32],(const float*)d_in[36]};
  const float* bhh[4]={(const float*)d_in[25],(const float*)d_in[29],(const float*)d_in[33],(const float*)d_in[37]};
  const float* lpw=(const float*)d_in[38]; const float* lpb=(const float*)d_in[39];
  const float* nfpw=(const float*)d_in[40];
  const float* wfc=(const float*)d_in[41], *wffw=(const float*)d_in[42], *wffb=(const float*)d_in[43];
  const float* wasrc=(const float*)d_in[44], *wafeat=(const float*)d_in[45];
  const float* wlng=(const float*)d_in[46], *wlnb=(const float*)d_in[47];
  const float* ww1=(const float*)d_in[48], *wb1=(const float*)d_in[49];
  const float* ww2=(const float*)d_in[50], *wb2=(const float*)d_in[51];
  const float* sfc=(const float*)d_in[52], *sffw=(const float*)d_in[53], *sffb=(const float*)d_in[54];
  const float* sasrc=(const float*)d_in[55], *safeat=(const float*)d_in[56];
  const float* slng=(const float*)d_in[57], *slnb=(const float*)d_in[58];
  const float* sw1=(const float*)d_in[59], *sb1=(const float*)d_in[60];
  const float* sw2=(const float*)d_in[61], *sb2=(const float*)d_in[62];
  const float* wh0=(const float*)d_in[63]; const float* wh1=(const float*)d_in[64];

  // ---- bump allocator over workspace
  char* base=(char*)d_ws; size_t off=0;
  auto alloc=[&](size_t bytes)->void*{ size_t a=(off+255)&~(size_t)255; off=a+bytes; return (void*)(base+a); };

  h16*  tok     = (h16*)alloc((size_t)(NSENTS*100+32)*EDIM*2);      // 192MB, overlaid later
  float* hsum_w = (float*)tok;                                      // 96MB overlay (post-CNN)
  h16*  mid_w   = (h16*)((char*)tok + 96000000);                    // 82MB overlay

  h16*  convt   = (h16*)alloc(524288*2);
  h16*  cnnpt   = (h16*)alloc((size_t)128*320*2);
  h16*  wih_t[4]; 
  wih_t[0]=(h16*)alloc((size_t)512*320*2); wih_t[1]=(h16*)alloc((size_t)512*320*2);
  wih_t[2]=(h16*)alloc((size_t)512*256*2); wih_t[3]=(h16*)alloc((size_t)512*256*2);
  h16*  whh_t[4]; for(int i=0;i<4;++i) whh_t[i]=(h16*)alloc((size_t)512*128*2);
  float* biasc[4]; for(int i=0;i<4;++i) biasc[i]=(float*)alloc(512*4);
  h16*  lpt     = (h16*)alloc((size_t)128*256*2);
  h16*  nfpt    = (h16*)alloc((size_t)64*256*2);
  h16*  fcw_t   = (h16*)alloc((size_t)64*320*2);
  h16*  fcs_t   = (h16*)alloc((size_t)320*64*2);
  h16*  w1t_w   = (h16*)alloc((size_t)512*64*2);
  h16*  w2t_w   = (h16*)alloc((size_t)64*512*2);
  h16*  w1t_s   = (h16*)alloc((size_t)512*320*2);
  h16*  w2t_s   = (h16*)alloc((size_t)320*512*2);
  float* dfs_w  = (float*)alloc(80*4);
  float* dfs_s  = (float*)alloc(60*4);
  float* cbp    = (float*)alloc(6*64*4);

  h16*  ngram_h = (h16*)alloc((size_t)NSENTS*320*2);
  h16*  ngramp_h= (h16*)alloc((size_t)NSENTS*320*2);
  float* preA   = (float*)alloc((size_t)NSENTS*512*4);
  float* preB   = (float*)alloc((size_t)NSENTS*512*4);
  h16*  x2a     = (h16*)alloc((size_t)NSENTS*256*2);
  h16*  x2b     = (h16*)alloc((size_t)NSENTS*256*2);
  h16*  sentcat = (h16*)alloc((size_t)NSENTS*256*2);
  float* sfeat  = (float*)alloc((size_t)NSENTS*64*4);
  h16*  wfh     = (h16*)alloc((size_t)NWORDS*320*2);     // word feat / y_w / new word state (f16)
  float* wstate = (float*)alloc((size_t)NWORDS*EDIM*4);  // word state f32
  float* z_w    = (float*)alloc((size_t)NWORDS*64*4);
  float* z_s    = (float*)alloc((size_t)NSENTS*EDIM*4);
  float* sal_w  = (float*)alloc((size_t)NWORDS*8*4);
  float* sal_s  = (float*)alloc((size_t)NSENTS*6*4);
  unsigned* amax_s = (unsigned*)alloc((size_t)NSENTS*8*4);
  float* den_s  = (float*)alloc((size_t)NSENTS*8*4);
  float* hsum_s = (float*)alloc((size_t)NSENTS*64*4);
  unsigned* amax_w = (unsigned*)alloc((size_t)NWORDS*6*4);
  float* den_w  = (float*)alloc((size_t)NWORDS*6*4);
  float* att_e  = (float*)alloc((size_t)NEDGES*8*4);
  float* ex_e   = (float*)alloc((size_t)NEDGES*8*4);
  h16*  yh_s    = (h16*)alloc((size_t)NSENTS*64*2);
  h16*  mid_s   = (h16*)alloc((size_t)NSENTS*512*2);
  float* sstate = (float*)alloc((size_t)NSENTS*64*4);
  h16*  sstate_h= (h16*)alloc((size_t)NSENTS*64*2);

  auto gemm=[&](const h16* A,int lda,const h16* Bt,int ldb,float* C,int ldc,h16* Ch,int ldh,
                const float* bias,const float* res,int M,int N,int Kp,int relu){
    dim3 g((unsigned)((N+63)/64),(unsigned)((M+127)/128));
    gemm_h<<<g,256,0,stream>>>(A,lda,Bt,ldb,C,ldc,Ch,ldh,bias,res,M,N,Kp/32,relu);
  };
  auto zero=[&](void* p,long words){
    zero_u32<<<dim3((unsigned)((words+255)/256)),256,0,stream>>>((unsigned*)p,words);
  };
  auto tconv=[&](const float* W,h16* Bt,int K,int N,int Kp,int Np){
    long t=(long)Np*Kp; conv_trans<<<dim3((unsigned)((t+255)/256)),256,0,stream>>>(W,Bt,K,N,Kp,Np);
  };
  auto trow=[&](const float* W,h16* Bt,int N,int K,int Kp,int Np){
    long t=(long)Np*Kp; copy_rowmajor<<<dim3((unsigned)((t+255)/256)),256,0,stream>>>(W,Bt,N,K,Kp,Np);
  };

  // ---- parameter prep
  {
    const int kst[6]={19,29,38,47,57,66}; const int koff[6]={0,38912,98304,176128,272384,389120};
    for (int i=0;i<6;++i){
      tconv(cvw[i], convt+koff[i], (i+2)*300, 50, kst[i]*32, 64);
      copy_pad_f<<<1,64,0,stream>>>(cvb[i], cbp+i*64, 50, 64);
    }
  }
  tconv(cnnw, cnnpt, 300, 128, 320, 128);
  trow(Wih[0], wih_t[0], 512, 300, 320, 512);
  trow(Wih[1], wih_t[1], 512, 300, 320, 512);
  trow(Wih[2], wih_t[2], 512, 256, 256, 512);
  trow(Wih[3], wih_t[3], 512, 256, 256, 512);
  for (int i=0;i<4;++i){
    trow(Whh[i], whh_t[i], 512, 128, 128, 512);
    addb<<<2,256,0,stream>>>(bih[i], bhh[i], biasc[i], 512);
  }
  tconv(lpw, lpt, 256, 128, 256, 128);
  tconv(nfpw, nfpt, 256, 64, 256, 64);
  {
    long t=(long)64*320; fc_trans<<<dim3((unsigned)((t+255)/256)),256,0,stream>>>(wfc, fcw_t, 8, 300, 8, 320, 64);
    t=(long)320*64;      fc_trans<<<dim3((unsigned)((t+255)/256)),256,0,stream>>>(sfc, fcs_t, 6, 64, 50, 64, 320);
  }
  tconv(ww1, w1t_w, 64, 512, 64, 512);
  tconv(ww2, w2t_w, 512, 64, 512, 64);
  tconv(sw1, w1t_s, 300, 512, 320, 512);
  tconv(sw2, w2t_s, 512, 300, 512, 320);
  gat_dfs<<<1,128,0,stream>>>(tfe, wffw, wffb, wafeat, dfs_w, 8, 8);
  gat_dfs<<<1,128,0,stream>>>(tfe, sffw, sffb, safeat, dfs_s, 6, 50);

  // ---- features
  {
    long tot=(long)(NSENTS*100+32)*EDIM;
    build_tok<<<dim3((unsigned)((tot+255)/256)),256,0,stream>>>(sw, embed, tok, tot);
    long wtot=(long)NWORDS*320;
    build_wordf<<<dim3((unsigned)((wtot+255)/256)),256,0,stream>>>(wid, embed, wfh, wstate);
  }
  conv_ngram<<<NSENTS,256,0,stream>>>(tok, convt, cbp, spos, ngram_h, ngramp_h);

  // cnn_feat -> sentcat[:,0:128]
  gemm(ngramp_h,320, cnnpt,320, nullptr,0, sentcat,256, cnnb, nullptr, NSENTS,128,320, 0);
  // BiLSTM layer1
  gemm(ngram_h,320, wih_t[0],320, preA,512, nullptr,0, biasc[0], nullptr, NSENTS,512,320, 0);
  gemm(ngram_h,320, wih_t[1],320, preB,512, nullptr,0, biasc[1], nullptr, NSENTS,512,320, 0);
  lstm_scan<<<1,512,0,stream>>>(preA, whh_t[0], x2a, 256, 0,   0);
  lstm_scan<<<1,512,0,stream>>>(preB, whh_t[1], x2a, 256, 128, 1);
  // BiLSTM layer2
  gemm(x2a,256, wih_t[2],256, preA,512, nullptr,0, biasc[2], nullptr, NSENTS,512,256, 0);
  gemm(x2a,256, wih_t[3],256, preB,512, nullptr,0, biasc[3], nullptr, NSENTS,512,256, 0);
  lstm_scan<<<1,512,0,stream>>>(preA, whh_t[2], x2b, 256, 0,   0);
  lstm_scan<<<1,512,0,stream>>>(preB, whh_t[3], x2b, 256, 128, 1);
  // lstm_feat -> sentcat[:,128:256]
  gemm(x2b,256, lpt,256, nullptr,0, sentcat+128,256, lpb, nullptr, NSENTS,128,256, 0);
  // sent_feature
  gemm(sentcat,256, nfpt,256, sfeat,64, nullptr,0, nullptr, nullptr, NSENTS,64,256, 0);

  const int eg = (NEDGES+255)/256;
  // ---- GAT block runner
  auto run_gat=[&](const h16* srcA,int lda,const h16* fcBt,int fcKp,
                   float* z,int nsrc,int nh,int dh,
                   const int* es,const int* ed,float* sal,const float* dfs,
                   unsigned* amax,float* den,float* hsum,int ndst,int dim,int kpad,
                   const float* origin,const float* lng,const float* lnb,
                   h16* yh,const h16* w1t,const float* b1,h16* mid,
                   const h16* w2t,const float* b2,float* outF,h16* outH){
    gemm(srcA,lda, fcBt,fcKp, z,dim, nullptr,0, nullptr, nullptr, nsrc,dim,fcKp, 0);
    long sn=(long)nsrc*nh;
    gat_salpha<<<dim3((unsigned)((sn+255)/256)),256,0,stream>>>(z, (nh==8)?wasrc:sasrc, sal, nsrc, nh, dh);
    zero(amax,(long)ndst*nh); zero(den,(long)ndst*nh); zero(hsum,(long)ndst*dim);
    gat_e1<<<eg,256,0,stream>>>(es,ed,tff,sal,dfs,att_e,amax,NEDGES,nh);
    gat_e2<<<eg,256,0,stream>>>(ed,att_e,amax,ex_e,den,NEDGES,nh);
    long e3=(long)NEDGES*dim;
    gat_e3<<<dim3((unsigned)((e3+255)/256)),256,0,stream>>>(es,ed,ex_e,den,z,hsum,NEDGES,nh,dh,dim);
    gat_node<<<dim3((unsigned)((ndst+7)/8)),256,0,stream>>>(hsum,origin,lng,lnb,yh,dim,kpad,ndst);
    gemm(yh,kpad, w1t,kpad, nullptr,0, mid,512, b1, nullptr, ndst,512,kpad, 1);
    gemm(mid,512, w2t,512, outF,dim, outH,kpad, b2, hsum, ndst,dim,512, 0);
  };

  // w2s #1 : words -> sents, origin = sent_feature
  run_gat(wfh,320, fcw_t,320, z_w, NWORDS,8,8, esrc,edst, sal_w,dfs_w,
          amax_s,den_s,hsum_s, NSENTS,64,64, sfeat,wlng,wlnb,
          yh_s,w1t_w,wb1,mid_s, w2t_w,wb2, sstate,sstate_h);
  // s2w : sents -> words, origin = word_feat (in wstate)
  run_gat(sstate_h,64, fcs_t,64, z_s, NSENTS,6,50, edst,esrc, sal_s,dfs_s,
          amax_w,den_w,hsum_w, NWORDS,300,320, wstate,slng,slnb,
          wfh,w1t_s,sb1,mid_w, w2t_s,sb2, wstate,wfh);
  // w2s #2 : new words -> sents, origin = previous sent_state
  run_gat(wfh,320, fcw_t,320, z_w, NWORDS,8,8, esrc,edst, sal_w,dfs_w,
          amax_s,den_s,hsum_s, NSENTS,64,64, sstate,wlng,wlnb,
          yh_s,w1t_w,wb1,mid_s, w2t_w,wb2, sstate,nullptr);

  final_out<<<dim3((NSENTS*2+255)/256),256,0,stream>>>(sstate, wh0, wh1, (float*)d_out);
  (void)in_sizes; (void)out_size; (void)ws_size;
}
